// DotProductAttention_43267500540680
// MI455X (gfx1250) — matrix operand order repaired and checked by reading, not compiled, not run on hardware
//
#include <hip/hip_runtime.h>
#include <type_traits>

typedef __attribute__((ext_vector_type(16))) __bf16 v16bf;
typedef __attribute__((ext_vector_type(8)))  __bf16 v8bf;
typedef __attribute__((ext_vector_type(8)))  float  v8f;
typedef __attribute__((ext_vector_type(4)))  float  v4f;

union BF16x16 { v16bf v; v8bf h[2]; };
union PKBF    { __bf16 h[2]; unsigned u; };

// Problem constants (reference: B=64, S=1024, D=64)
constexpr int kS  = 1024;
constexpr int kD  = 64;
constexpr int kBQ = 128;   // queries per block (8 waves x 16)
constexpr int kTQ = 16;    // queries per wave
constexpr int kTK = 32;    // keys per flash iteration
constexpr int kWAVES = 8;

// log2(e)/sqrt(D): folds the attention scale and exp->exp2 conversion into Q
// once (softmax is invariant under a positive rescale of the logit base).
#define QSCALE (0.125f * 1.44269504088896340736f)

// max without canonicalize ops (inputs are never sNaN here).
__device__ __forceinline__ float vmax(float a, float b) {
    float d;
    asm("v_max_num_f32 %0, %1, %2" : "=v"(d) : "v"(a), "v"(b));
    return d;
}

// Full 16-lane row max: 4 fused max+DPP ROW_XMASK steps, 1 VALU op each.
__device__ __forceinline__ float row_max16(float x) {
    asm("v_max_num_f32_dpp %0, %0, %0 row_xmask:1 row_mask:0xf bank_mask:0xf bound_ctrl:1\n\t"
        "v_max_num_f32_dpp %0, %0, %0 row_xmask:2 row_mask:0xf bank_mask:0xf bound_ctrl:1\n\t"
        "v_max_num_f32_dpp %0, %0, %0 row_xmask:4 row_mask:0xf bank_mask:0xf bound_ctrl:1\n\t"
        "v_max_num_f32_dpp %0, %0, %0 row_xmask:8 row_mask:0xf bank_mask:0xf bound_ctrl:1"
        : "+v"(x));
    return x;
}

__global__ __launch_bounds__(256)
void attn_flash_bf16_wmma(const float* __restrict__ Q,
                          const float* __restrict__ K,
                          const float* __restrict__ V,
                          const long long* __restrict__ vlens,
                          float* __restrict__ Out)
{
    // Double-buffered K/V tiles + per-wave P tile.
    // ldsVT / ldsP use the permuted key order j = 2*(k%16) + (k/16); the PV
    // contraction is invariant since P columns and V^T rows share the order.
    __shared__ __align__(16) __bf16 ldsK [2][kTK][kD];      // [buf][key][d]
    __shared__ __align__(16) __bf16 ldsVT[2][kD][kTK];      // [buf][d][j]
    __shared__ __align__(16) __bf16 ldsP [kWAVES][kTQ][kTK];// [wave][q][j]

    const int b    = blockIdx.y;
    const int tid  = threadIdx.x;
    const int wave = tid >> 5;
    const int lane = tid & 31;
    const int l16  = lane & 15;
    const int hi   = lane >> 4;          // 0: lanes 0-15, 1: lanes 16-31

    const long long vlen = vlens[b];
    const int  nfull   = (int)(vlen >> 5);          // tiles needing no mask
    const bool partial = (vlen & 31) != 0;
    const int  ntiles  = nfull + (partial ? 1 : 0);

    const int qbase = blockIdx.x * kBQ + wave * kTQ;
    const int qrow  = qbase + l16;

    // ---- Q fragments: A-layout (16x32 bf16), QSCALE folded in ----
    v16bf qa[2];
    {
        const float* qp = Q + ((size_t)b * kS + qrow) * kD;
        #pragma unroll
        for (int c = 0; c < 2; ++c) {
            const int d0 = c * 32 + hi * 8;
            v4f f0 = *(const v4f*)(qp + d0);
            v4f f1 = *(const v4f*)(qp + d0 + 4);
            v4f f2 = *(const v4f*)(qp + d0 + 16);
            v4f f3 = *(const v4f*)(qp + d0 + 20);
            BF16x16 t;
            #pragma unroll
            for (int i = 0; i < 4; ++i) {
                t.v[i]      = (__bf16)(f0[i] * QSCALE);
                t.v[4 + i]  = (__bf16)(f1[i] * QSCALE);
                t.v[8 + i]  = (__bf16)(f2[i] * QSCALE);
                t.v[12 + i] = (__bf16)(f3[i] * QSCALE);
            }
            qa[c] = t.v;
        }
    }

    // All-ones bf16 B matrix for WMMA row-sum.
    BF16x16 ones;
    #pragma unroll
    for (int i = 0; i < 16; ++i) ones.v[i] = (__bf16)1.0f;

    v8f acc[4];
    #pragma unroll
    for (int t = 0; t < 4; ++t) acc[t] = v8f{};
    float m_run[8], l_run[8];
    #pragma unroll
    for (int v = 0; v < 8; ++v) { m_run[v] = -1e30f; l_run[v] = 0.0f; }

    // Cooperative staging decomposition: 256 threads cover 32 keys x 64 dims.
    const int krow = tid >> 3;                         // 0..31
    const int d0   = (tid & 7) * 8;                    // 0..56
    const int jcol = ((krow & 15) << 1) | (krow >> 4); // permuted key column
    const float* kbasep = K + ((size_t)b * kS + krow) * kD + d0;
    const float* vbasep = V + ((size_t)b * kS + krow) * kD + d0;

    // Software-pipeline registers holding the next tile (global -> reg).
    v4f kr0, kr1, wr0, wr1;
    if (ntiles > 0) {
        kr0 = *(const v4f*)(kbasep);
        kr1 = *(const v4f*)(kbasep + 4);
        wr0 = *(const v4f*)(vbasep);
        wr1 = *(const v4f*)(vbasep + 4);
    }

    auto body = [&](auto masked_c, int it) {
        constexpr bool MASKED = decltype(masked_c)::value;
        const int kbase = it * kTK;
        const int buf   = it & 1;

        // ---- reg -> LDS store of tile `it` ----
        {
            v8bf kb;
            #pragma unroll
            for (int i = 0; i < 4; ++i) { kb[i] = (__bf16)kr0[i]; kb[4 + i] = (__bf16)kr1[i]; }
            *(v8bf*)&ldsK[buf][krow][d0] = kb;
            #pragma unroll
            for (int i = 0; i < 4; ++i) ldsVT[buf][d0 + i][jcol]     = (__bf16)wr0[i];
            #pragma unroll
            for (int i = 0; i < 4; ++i) ldsVT[buf][d0 + 4 + i][jcol] = (__bf16)wr1[i];
        }
        __syncthreads();   // tile `it` visible; prior reads of this buf complete

        // ---- issue global loads for tile `it+1` (overlap with compute) ----
        if (it + 1 < ntiles) {
            const float* kp = kbasep + (size_t)(it + 1) * kTK * kD;
            const float* vp = vbasep + (size_t)(it + 1) * kTK * kD;
            kr0 = *(const v4f*)(kp);
            kr1 = *(const v4f*)(kp + 4);
            wr0 = *(const v4f*)(vp);
            wr1 = *(const v4f*)(vp + 4);
            if (it + 2 < ntiles) {    // pull tile it+2 toward L2 (global_prefetch_b8)
                __builtin_prefetch(kp + kTK * kD, 0, 1);
                __builtin_prefetch(vp + kTK * kD, 0, 1);
            }
        }

        // ---- preload ALL four K B-fragments, then the score WMMA chain ----
        BF16x16 kb[2][2];
        #pragma unroll
        for (int h = 0; h < 2; ++h)
            #pragma unroll
            for (int c = 0; c < 2; ++c) {
                const int kr     = h * 16 + l16;
                const int dstart = c * 32 + hi * 8;
                kb[h][c].h[0] = *(const v8bf*)&ldsK[buf][kr][dstart];
                kb[h][c].h[1] = *(const v8bf*)&ldsK[buf][kr][dstart + 16];
            }
        v8f s[2];
        #pragma unroll
        for (int h = 0; h < 2; ++h) {
            v8f cacc = __builtin_amdgcn_wmma_f32_16x16x32_bf16(
                false, qa[0], false, kb[h][0].v, (short)0, v8f{}, false, false);
            s[h] = __builtin_amdgcn_wmma_f32_16x16x32_bf16(
                false, qa[1], false, kb[h][1].v, (short)0, cacc, false, false);
        }

        // ---- online softmax in exp2 domain: fused DPP row-max, v_exp, P -> LDS ----
        bool ok0 = true, ok1 = true;
        if constexpr (MASKED) {
            ok0 = (long long)(kbase + l16)      < vlen;
            ok1 = (long long)(kbase + 16 + l16) < vlen;
        }
        float alpha[8];
        #pragma unroll
        for (int v = 0; v < 8; ++v) {
            const float sc0 = s[0][v];
            const float sc1 = s[1][v];
            float mx;
            if constexpr (MASKED)
                mx = vmax(ok0 ? sc0 : -1e30f, ok1 ? sc1 : -1e30f);
            else
                mx = vmax(sc0, sc1);
            mx = row_max16(mx);
            const float m_new = vmax(m_run[v], mx);
            alpha[v] = __builtin_amdgcn_exp2f(m_run[v] - m_new);
            m_run[v] = m_new;
            float p0 = __builtin_amdgcn_exp2f(sc0 - m_new);
            float p1 = __builtin_amdgcn_exp2f(sc1 - m_new);
            if constexpr (MASKED) {
                p0 = ok0 ? p0 : 0.0f;
                p1 = ok1 ? p1 : 0.0f;
            }
            // packed store: columns j=2*l16 (key l16) and j=2*l16+1 (key 16+l16)
            PKBF pk;
            pk.h[0] = (__bf16)p0;
            pk.h[1] = (__bf16)p1;
            const int row = v + hi * 8;               // C layout row
            *(unsigned*)&ldsP[wave][row][l16 * 2] = pk.u;
        }

        // ---- preload P (A-layout) and ALL four V B-fragments ----
        BF16x16 pa;
        {
            const int j0 = hi * 8;
            pa.h[0] = *(const v8bf*)&ldsP[wave][l16][j0];
            pa.h[1] = *(const v8bf*)&ldsP[wave][l16][j0 + 16];
        }
        BF16x16 vb[4];
        #pragma unroll
        for (int t = 0; t < 4; ++t) {
            const int drow   = t * 16 + l16;
            const int jstart = hi * 8;
            vb[t].h[0] = *(const v8bf*)&ldsVT[buf][drow][jstart];
            vb[t].h[1] = *(const v8bf*)&ldsVT[buf][drow][jstart + 16];
        }
        #pragma unroll
        for (int t = 0; t < 4; ++t)
            #pragma unroll
            for (int v = 0; v < 8; ++v)
                acc[t][v] *= alpha[v];

        // ---- row sums via WMMA against ones: l = l*alpha + rowsum(P) ----
        {
            v8f rsum = __builtin_amdgcn_wmma_f32_16x16x32_bf16(
                false, pa.v, false, ones.v, (short)0, v8f{}, false, false);
            #pragma unroll
            for (int v = 0; v < 8; ++v)
                l_run[v] = l_run[v] * alpha[v] + rsum[v];
        }

        // ---- O += P * V over four 16-wide d tiles (permuted key order) ----
        #pragma unroll
        for (int t = 0; t < 4; ++t)
            acc[t] = __builtin_amdgcn_wmma_f32_16x16x32_bf16(
                false, pa.v, false, vb[t].v, (short)0, acc[t], false, false);

        // ---- desired pipeline for this scheduling region ----
        // (masks: 0x008 WMMA, 0x100 DS-read, 0x200 DS-write)
        __builtin_amdgcn_sched_group_barrier(0x100,  8, 0); // K frag ds_reads
        __builtin_amdgcn_sched_group_barrier(0x008,  4, 0); // score WMMAs
        __builtin_amdgcn_sched_group_barrier(0x200,  8, 0); // P ds_writes
        __builtin_amdgcn_sched_group_barrier(0x100, 10, 0); // P + V ds_reads
        __builtin_amdgcn_sched_group_barrier(0x008,  5, 0); // rowsum + PV WMMAs
    };

    for (int it = 0; it < nfull; ++it)
        body(std::integral_constant<bool, false>{}, it);
    if (partial)
        body(std::integral_constant<bool, true>{}, nfull);

    // ---- finalize: O = acc / l (0 where the whole row is masked) ----
    #pragma unroll
    for (int t = 0; t < 4; ++t) {
        #pragma unroll
        for (int v = 0; v < 8; ++v) {
            const int row = v + hi * 8;
            const float l = l_run[v];
            const float o = (l > 0.0f) ? acc[t][v] / l : 0.0f;
            Out[((size_t)b * kS + qbase + row) * kD + t * 16 + l16] = o;
        }
    }
}

extern "C" void kernel_launch(void* const* d_in, const int* in_sizes, int n_in,
                              void* d_out, int out_size, void* d_ws, size_t ws_size,
                              hipStream_t stream) {
    (void)in_sizes; (void)n_in; (void)out_size; (void)d_ws; (void)ws_size;
    const float*     Q  = (const float*)d_in[0];
    const float*     K  = (const float*)d_in[1];
    const float*     V  = (const float*)d_in[2];
    const long long* VL = (const long long*)d_in[3];
    float* O = (float*)d_out;

    dim3 grid(kS / kBQ, 64);   // (query blocks, batch)
    dim3 block(256);
    attn_flash_bf16_wmma<<<grid, block, 0, stream>>>(Q, K, V, VL, O);
}
